// Global_ATT_11501922419472
// MI455X (gfx1250) — matrix operand order repaired and checked by reading, not compiled, not run on hardware
//
#include <hip/hip_runtime.h>
#include <hip/hip_bf16.h>
#include <math.h>

#define HIDDEN 256
#define HEADS  8
#define DHEAD  32
#define NTOK   4096

typedef __attribute__((ext_vector_type(16))) __bf16 v16bf;
typedef __attribute__((ext_vector_type(8)))  __bf16 v8bf;
typedef __attribute__((ext_vector_type(8)))  float  v8f;
typedef __attribute__((ext_vector_type(2)))  int    v2i;

#if defined(__has_builtin)
#  if __has_builtin(__builtin_amdgcn_global_load_async_to_lds_b64)
#    define HAVE_ASYNC_LDS 1
#  endif
#endif
#ifndef HAVE_ASYNC_LDS
#  define HAVE_ASYNC_LDS 0
#endif

#if HAVE_ASYNC_LDS
// per hipcc diagnostic: param0 is v2i* in global (AS1); param1 is the LDS side (AS3)
typedef __attribute__((address_space(1))) v2i* gbuf_p;
typedef __attribute__((address_space(3))) v2i* lbuf_p;
#endif

// round-to-nearest-even f32 -> bf16 (bit ops only)
static __device__ inline __bf16 f2bf(float f) {
    unsigned u = __builtin_bit_cast(unsigned, f);
    unsigned r = (u + 0x7FFFu + ((u >> 16) & 1u)) >> 16;
    unsigned short s = (unsigned short)r;
    return __builtin_bit_cast(__bf16, s);
}

static __device__ inline v16bf cat8(v8bf lo, v8bf hi) {
    return __builtin_shufflevector(lo, hi, 0,1,2,3,4,5,6,7,8,9,10,11,12,13,14,15);
}

// A-fragment 16x32 bf16 (ISA 7.12.2): lane l holds row (l&15);
// K chunks [8*half, +8) and [16+8*half, +8), half = l>>4.
static __device__ inline v16bf load_afrag(const __bf16* src, int ldm, int row0, int k0, int lane) {
    int half = lane >> 4, lr = lane & 15;
    const __bf16* p = src + (size_t)(row0 + lr) * ldm + k0;
    v8bf lo = *(const v8bf*)(p + 8 * half);
    v8bf hi = *(const v8bf*)(p + 16 + 8 * half);
    return cat8(lo, hi);
}

// B-fragment 32x16 bf16: lane l holds column (l&15); K chunks
// [8*half, +8) and [16+8*half, +8). src rows indexed by N, contiguous in K.
static __device__ inline v16bf load_bfrag(const __bf16* src, int ldm, int row0, int k0, int lane) {
    int half = lane >> 4, lr = lane & 15;
    const __bf16* p = src + (size_t)(row0 + lr) * ldm + k0;
    v8bf lo = *(const v8bf*)(p + 8 * half);
    v8bf hi = *(const v8bf*)(p + 16 + 8 * half);
    return cat8(lo, hi);
}

static __device__ inline v8f wmma_bf16(v16bf a, v16bf b, v8f c) {
    return __builtin_amdgcn_wmma_f32_16x16x32_bf16(false, a, false, b, (short)0, c, false, false);
}

__global__ void cvt_f32_bf16(const float* __restrict__ in, __bf16* __restrict__ out, int n) {
    int i = blockIdx.x * blockDim.x + threadIdx.x;
    if (i < n) out[i] = f2bf(in[i]);
}

// Q/K/V projections: out = X @ W^T.  Q,K stored [h][n][d]; V transposed [h][d][n].
__global__ void proj_qkv(const __bf16* __restrict__ Xb,
                         const __bf16* __restrict__ Wq,
                         const __bf16* __restrict__ Wk,
                         const __bf16* __restrict__ Wv,
                         __bf16* __restrict__ Q,
                         __bf16* __restrict__ K,
                         __bf16* __restrict__ Vt) {
    int lane = threadIdx.x & 31, wid = threadIdx.x >> 5;
    int it = blockIdx.x * 4 + wid;
    int jt = blockIdx.y;
    int wsel = blockIdx.z;
    const __bf16* W = (wsel == 0) ? Wq : ((wsel == 1) ? Wk : Wv);
    int i0 = it * 16, j0 = jt * 16;

    v8f acc = {};
    #pragma unroll
    for (int k0 = 0; k0 < HIDDEN; k0 += 32) {
        v16bf a = load_afrag(Xb, HIDDEN, i0, k0, lane);
        v16bf b = load_bfrag(W,  HIDDEN, j0, k0, lane);
        acc = wmma_bf16(a, b, acc);
    }
    int half = lane >> 4, lr = lane & 15;
    int h = j0 >> 5;
    int dcol = (j0 & 31) + lr;
    #pragma unroll
    for (int r = 0; r < 8; ++r) {
        int n = i0 + r + 8 * half;
        __bf16 v = f2bf(acc[r]);
        if (wsel == 0)      Q[((size_t)(h * NTOK + n)) * DHEAD + dcol] = v;
        else if (wsel == 1) K[((size_t)(h * NTOK + n)) * DHEAD + dcol] = v;
        else                Vt[((size_t)(h * DHEAD + dcol)) * NTOK + n] = v;
    }
}

// Cooperative fill of one 32-key K tile (32x32 bf16, contiguous 2KB) and one
// V tile (32 head-dims x 32 keys from the transposed V) into LDS.
// 256 threads x 8B each per tile; async (ASYNCcnt) when available.
static __device__ inline void fill_kv(const __bf16* __restrict__ Kh,
                                      const __bf16* __restrict__ Vh,
                                      __bf16* lk, __bf16* lv, int m0, int tid) {
    const __bf16* gk = Kh + (size_t)m0 * DHEAD + tid * 4;   // contiguous block
    int d = tid >> 3, mc = (tid & 7) * 4;
    const __bf16* gv = Vh + (size_t)d * NTOK + m0 + mc;
#if HAVE_ASYNC_LDS
    __builtin_amdgcn_global_load_async_to_lds_b64(
        (gbuf_p)(void*)gk, (lbuf_p)(void*)(lk + tid * 4), 0, 0);
    __builtin_amdgcn_global_load_async_to_lds_b64(
        (gbuf_p)(void*)gv, (lbuf_p)(void*)(lv + d * 32 + mc), 0, 0);
#else
    *(uint64_t*)(lk + tid * 4)     = *(const uint64_t*)gk;
    *(uint64_t*)(lv + d * 32 + mc) = *(const uint64_t*)gv;
#endif
}

// Fused attention. Block = 8 waves sharing one head; K/V tiles double-buffered
// in LDS (8x reduction of global/L2 reads vs per-wave streaming).
__global__ void attn(const __bf16* __restrict__ Q,
                     const __bf16* __restrict__ K,
                     const __bf16* __restrict__ Vt,
                     __bf16* __restrict__ attb) {
    __shared__ __bf16 kbuf[2][32 * 32];     // 2 x 2KB
    __shared__ __bf16 vbuf[2][32 * 32];     // 2 x 2KB
    __shared__ __bf16 stage[8][16][40];     // per-wave score staging, 80B rows

    int tid  = threadIdx.x;
    int lane = tid & 31, wid = tid >> 5;
    int it = blockIdx.x * 8 + wid;          // 0..255 token tile
    int h  = blockIdx.y;                    // 0..7
    int i0 = it * 16;
    int half = lane >> 4, lr = lane & 15;

    const __bf16* Qh = Q  + (size_t)h * NTOK * DHEAD;
    const __bf16* Kh = K  + (size_t)h * NTOK * DHEAD;
    const __bf16* Vh = Vt + (size_t)h * DHEAD * NTOK;

    v16bf qf = load_afrag(Qh, DHEAD, i0, 0, lane);   // reused for all 128 steps
    v8f acc0 = {}, acc1 = {};
    v8f zero = {};

    const int NIT = NTOK / 32;              // 128
    fill_kv(Kh, Vh, kbuf[0], vbuf[0], 0, tid);

    for (int i = 0; i < NIT; ++i) {
        bool have_next = (i + 1) < NIT;
        if (have_next)
            fill_kv(Kh, Vh, kbuf[(i + 1) & 1], vbuf[(i + 1) & 1], (i + 1) * 32, tid);
#if HAVE_ASYNC_LDS
        // async loads complete in order: <=2 outstanding means tile i landed
        if (have_next) asm volatile("s_wait_asynccnt 0x2" ::: "memory");
        else           asm volatile("s_wait_asynccnt 0x0" ::: "memory");
#endif
        __syncthreads();                    // tile i visible to all waves

        const __bf16* lk = kbuf[i & 1];
        const __bf16* lv = vbuf[i & 1];

        v16bf kf0 = load_bfrag(lk, 32, 0,  0, lane);
        v16bf kf1 = load_bfrag(lk, 32, 16, 0, lane);
        v8f s0 = wmma_bf16(qf, kf0, zero);
        v8f s1 = wmma_bf16(qf, kf1, zero);

        // tanh + C-layout -> LDS (row = r + 8*half, col = lr / 16+lr)
        #pragma unroll
        for (int r = 0; r < 8; ++r) {
            int row = r + 8 * half;
            stage[wid][row][lr]      = f2bf(tanhf(s0[r]));
            stage[wid][row][16 + lr] = f2bf(tanhf(s1[r]));
        }
        asm volatile("s_wait_dscnt 0x0" ::: "memory");   // cross-lane RAW in wave

        v16bf sf  = load_afrag(&stage[wid][0][0], 40, 0, 0, lane);
        v16bf vf0 = load_bfrag(lv, 32, 0,  0, lane);
        v16bf vf1 = load_bfrag(lv, 32, 16, 0, lane);
        acc0 = wmma_bf16(sf, vf0, acc0);
        acc1 = wmma_bf16(sf, vf1, acc1);

        __syncthreads();                    // all waves done with tile i
    }

    #pragma unroll
    for (int r = 0; r < 8; ++r) {
        int n = i0 + r + 8 * half;
        attb[(size_t)n * HIDDEN + h * DHEAD + lr]      = f2bf(acc0[r]);
        attb[(size_t)n * HIDDEN + h * DHEAD + 16 + lr] = f2bf(acc1[r]);
    }
}

// att_hidden = relu(att @ Wf^T + bf), fp32 output
__global__ void final_proj(const __bf16* __restrict__ Ab,
                           const __bf16* __restrict__ Wf,
                           const float* __restrict__ bias,
                           float* __restrict__ out) {
    int lane = threadIdx.x & 31, wid = threadIdx.x >> 5;
    int it = blockIdx.x * 4 + wid;
    int jt = blockIdx.y;
    int i0 = it * 16, j0 = jt * 16;

    v8f acc = {};
    #pragma unroll
    for (int k0 = 0; k0 < HIDDEN; k0 += 32) {
        v16bf a = load_afrag(Ab, HIDDEN, i0, k0, lane);
        v16bf b = load_bfrag(Wf, HIDDEN, j0, k0, lane);
        acc = wmma_bf16(a, b, acc);
    }
    int half = lane >> 4, lr = lane & 15;
    float bj = bias[j0 + lr];
    #pragma unroll
    for (int r = 0; r < 8; ++r) {
        int n = i0 + r + 8 * half;
        float v = acc[r] + bj;
        out[(size_t)n * HIDDEN + j0 + lr] = v > 0.f ? v : 0.f;
    }
}

extern "C" void kernel_launch(void* const* d_in, const int* in_sizes, int n_in,
                              void* d_out, int out_size, void* d_ws, size_t ws_size,
                              hipStream_t stream) {
    const float* x  = (const float*)d_in[0];
    const float* wq = (const float*)d_in[1];
    const float* wk = (const float*)d_in[2];
    const float* wv = (const float*)d_in[3];
    const float* wf = (const float*)d_in[4];
    const float* bf = (const float*)d_in[5];
    float* out = (float*)d_out;

    char* ws = (char*)d_ws;
    __bf16* Xb   = (__bf16*)(ws);                                // 2 MB
    __bf16* Wqb  = (__bf16*)(ws + (2u << 20));                   // 128 KB each
    __bf16* Wkb  = Wqb + HIDDEN * HIDDEN;
    __bf16* Wvb  = Wkb + HIDDEN * HIDDEN;
    __bf16* Wfb  = Wvb + HIDDEN * HIDDEN;
    __bf16* Qb   = (__bf16*)(ws + (2u << 20) + (512u << 10));    // 2 MB each
    __bf16* Kb   = Qb  + (size_t)NTOK * HIDDEN;
    __bf16* Vtb  = Kb  + (size_t)NTOK * HIDDEN;
    __bf16* Attb = Vtb + (size_t)NTOK * HIDDEN;                  // ~10.5 MB total

    (void)hipMemcpyAsync(out, x, (size_t)NTOK * HIDDEN * sizeof(float),
                         hipMemcpyDeviceToDevice, stream);       // output[0] = inputs

    cvt_f32_bf16<<<(NTOK * HIDDEN + 255) / 256, 256, 0, stream>>>(x, Xb, NTOK * HIDDEN);
    cvt_f32_bf16<<<(HIDDEN * HIDDEN + 255) / 256, 256, 0, stream>>>(wq, Wqb, HIDDEN * HIDDEN);
    cvt_f32_bf16<<<(HIDDEN * HIDDEN + 255) / 256, 256, 0, stream>>>(wk, Wkb, HIDDEN * HIDDEN);
    cvt_f32_bf16<<<(HIDDEN * HIDDEN + 255) / 256, 256, 0, stream>>>(wv, Wvb, HIDDEN * HIDDEN);
    cvt_f32_bf16<<<(HIDDEN * HIDDEN + 255) / 256, 256, 0, stream>>>(wf, Wfb, HIDDEN * HIDDEN);

    proj_qkv<<<dim3(64, 16, 3), 128, 0, stream>>>(Xb, Wqb, Wkb, Wvb, Qb, Kb, Vtb);
    attn<<<dim3(32, 8), 256, 0, stream>>>(Qb, Kb, Vtb, Attb);
    final_proj<<<dim3(64, 16), 128, 0, stream>>>(Attb, Wfb, bf, out + (size_t)NTOK * HIDDEN);
}
